// ConditionalLinear_1623497638479
// MI455X (gfx1250) — compile-verified, matrix operand order checked by reading, not run on hardware
//
#include <hip/hip_runtime.h>
#include <hip/hip_bf16.h>

// CDNA5 / gfx1250: wave32, WMMA 16x16x32 bf16 (f32 accum), bf16x3 split GEMM.
typedef __attribute__((ext_vector_type(16))) __bf16 v16bf;
typedef __attribute__((ext_vector_type(8)))  float  v8f;

#define BDIM      1024            // N = FIN = FOUT
#define KC        32              // K-slab per iteration (one bf16 WMMA K)
#define TILE      128             // output tile per workgroup (M and N)
#define PITCH_DW  20              // LDS row pitch in dwords = (KC + 8 pad) halves
#define PLANE     (TILE * PITCH_DW)   // 2560 dwords per plane
// planes: [buf][ Xh | Xl | Wh | Wl ], double buffered

__device__ __forceinline__ void loadg(float4 r[4], const float* __restrict__ g) {
  const float4* p = (const float4*)g;
  r[0] = p[0]; r[1] = p[1]; r[2] = p[2]; r[3] = p[3];
}

// Truncation-based f32 -> (bf16 hi, bf16 lo) split; lo corrects hi's drop.
// Per-element-pair: 2 and, 2 sub, 2 shr, 1 and, 2 or  (~9 VALU).
__device__ __forceinline__ void cvt_store(const float4 r[4],
                                          unsigned int* __restrict__ shi,
                                          unsigned int* __restrict__ slo) {
  float f[16];
#pragma unroll
  for (int i = 0; i < 4; ++i) {
    f[4*i+0] = r[i].x; f[4*i+1] = r[i].y; f[4*i+2] = r[i].z; f[4*i+3] = r[i].w;
  }
  unsigned int hp[8], lp[8];
#pragma unroll
  for (int j = 0; j < 8; ++j) {
    const unsigned int ua = __float_as_uint(f[2*j]);
    const unsigned int uc = __float_as_uint(f[2*j+1]);
    const unsigned int uah = ua & 0xffff0000u;
    const unsigned int uch = uc & 0xffff0000u;
    hp[j] = (ua >> 16) | uch;
    const float ra = f[2*j]   - __uint_as_float(uah);
    const float rc = f[2*j+1] - __uint_as_float(uch);
    lp[j] = (__float_as_uint(ra) >> 16) | (__float_as_uint(rc) & 0xffff0000u);
  }
  *(uint4*)(shi + 0) = make_uint4(hp[0], hp[1], hp[2], hp[3]);
  *(uint4*)(shi + 4) = make_uint4(hp[4], hp[5], hp[6], hp[7]);
  *(uint4*)(slo + 0) = make_uint4(lp[0], lp[1], lp[2], lp[3]);
  *(uint4*)(slo + 4) = make_uint4(lp[4], lp[5], lp[6], lp[7]);
}

// 16 contiguous bf16 per lane; only 16B-aligned (pitch = 80B) -> two uint4s.
__device__ __forceinline__ v16bf load_frag(const unsigned int* __restrict__ lds,
                                           int dw) {
  union { uint4 q[2]; v16bf v; } u;
  u.q[0] = *(const uint4*)(lds + dw);
  u.q[1] = *(const uint4*)(lds + dw + 4);
  return u.v;
}

__global__ __launch_bounds__(256)
void cond_linear_bf16x3_kernel(const float* __restrict__ x,
                               const int*   __restrict__ task_id,
                               const float* __restrict__ W,
                               float*       __restrict__ out) {
  __shared__ unsigned int lds[2 * 4 * PLANE];   // 80 KB, double buffered

  const int b    = blockIdx.z;
  const int m0   = blockIdx.y * TILE;          // rows of X / out
  const int c0   = blockIdx.x * TILE;          // rows of W == cols of out
  const int task = task_id[b];

  const float* xb = x + ((size_t)b    << 20);  // X_b [1024,1024], K-contig
  const float* wb = W + ((size_t)task << 20);  // W_t [1024,1024], K-contig

  const int t    = threadIdx.x;
  const int lane = t & 31;
  const int wv   = t >> 5;                     // 8 waves
  const int wy   = wv >> 1;                    // 0..3 : 32-row strip of M
  const int wx   = wv & 1;                     // 0..1 : 64-col strip of N

  // staging: thread -> (row 0..127, 16-element half of the K-slab)
  const int srow   = t >> 1;
  const int shalf  = (t & 1) * 16;
  const int sdw    = srow * PITCH_DW + (t & 1) * 8;   // plane-relative dwords
  const float* gx  = xb + (size_t)(m0 + srow) * BDIM + shalf;
  const float* gw  = wb + (size_t)(c0 + srow) * BDIM + shalf;

  // fragment offsets (plane-relative dwords)
  const int khdw = (lane >> 4) * 8;
  int aoff[2], boff[4];
#pragma unroll
  for (int mi = 0; mi < 2; ++mi)
    aoff[mi] = (wy * 32 + mi * 16 + (lane & 15)) * PITCH_DW + khdw;
#pragma unroll
  for (int ni = 0; ni < 4; ++ni)
    boff[ni] = (wx * 64 + ni * 16 + (lane & 15)) * PITCH_DW + khdw;

  v8f acc[2][4];
#pragma unroll
  for (int mi = 0; mi < 2; ++mi)
#pragma unroll
    for (int ni = 0; ni < 4; ++ni)
      acc[mi][ni] = (v8f){0.f, 0.f, 0.f, 0.f, 0.f, 0.f, 0.f, 0.f};

  // ---- pipeline prologue: slab 0 into buffer 0 ----
  float4 rx[4], rw[4];
  loadg(rx, gx);
  loadg(rw, gw);
  {
    unsigned int* nb = lds;                    // buffer 0
    cvt_store(rx, nb + 0 * PLANE + sdw, nb + 1 * PLANE + sdw);
    cvt_store(rw, nb + 2 * PLANE + sdw, nb + 3 * PLANE + sdw);
  }
  __syncthreads();

  for (int k0 = 0; k0 < BDIM; k0 += KC) {
    const int  cur  = (k0 >> 5) & 1;
    const bool more = (k0 + KC) < BDIM;

    // issue next slab's global loads first; latency hides behind the WMMAs
    if (more) {
      loadg(rx, gx + k0 + KC);
      loadg(rw, gw + k0 + KC);
      if (k0 + 2 * KC < BDIM) {                // gfx1250 global_prefetch path
        __builtin_prefetch(gx + k0 + 2 * KC, 0, 3);
        __builtin_prefetch(gw + k0 + 2 * KC, 0, 3);
      }
    }

    const unsigned int* cb = lds + cur * 4 * PLANE;
    v16bf ah[2], al[2], bh[4], bl[4];
#pragma unroll
    for (int mi = 0; mi < 2; ++mi) {
      ah[mi] = load_frag(cb + 0 * PLANE, aoff[mi]);
      al[mi] = load_frag(cb + 1 * PLANE, aoff[mi]);
    }
#pragma unroll
    for (int ni = 0; ni < 4; ++ni) {
      bh[ni] = load_frag(cb + 2 * PLANE, boff[ni]);
      bl[ni] = load_frag(cb + 3 * PLANE, boff[ni]);
    }

    // bf16x3 split GEMM: lo*hi + hi*lo + hi*hi, f32 accumulate
#pragma unroll
    for (int mi = 0; mi < 2; ++mi)
#pragma unroll
      for (int ni = 0; ni < 4; ++ni) {
        acc[mi][ni] = __builtin_amdgcn_wmma_f32_16x16x32_bf16(
            false, al[mi], false, bh[ni], (short)0, acc[mi][ni], false, false);
        acc[mi][ni] = __builtin_amdgcn_wmma_f32_16x16x32_bf16(
            false, ah[mi], false, bl[ni], (short)0, acc[mi][ni], false, false);
        acc[mi][ni] = __builtin_amdgcn_wmma_f32_16x16x32_bf16(
            false, ah[mi], false, bh[ni], (short)0, acc[mi][ni], false, false);
      }

    // convert+store next slab into the other buffer; overlaps the XDL pipe
    if (more) {
      unsigned int* nb = lds + (cur ^ 1) * 4 * PLANE;
      cvt_store(rx, nb + 0 * PLANE + sdw, nb + 1 * PLANE + sdw);
      cvt_store(rw, nb + 2 * PLANE + sdw, nb + 3 * PLANE + sdw);
    }
    __syncthreads();
  }

  // Epilogue: C/D layout — VGPR j: M = j + 8*(lane>=16), N = lane%16
  const int  mrow = (lane >> 4) * 8;
  const int  ccol = lane & 15;
  float* ob = out + ((size_t)b << 20);
#pragma unroll
  for (int mi = 0; mi < 2; ++mi)
#pragma unroll
    for (int ni = 0; ni < 4; ++ni) {
      const int mbase = m0 + wy * 32 + mi * 16 + mrow;
      const int cbase = c0 + wx * 64 + ni * 16 + ccol;
#pragma unroll
      for (int j = 0; j < 8; ++j)
        ob[(size_t)(mbase + j) * BDIM + cbase] = acc[mi][ni][j];
    }

  // tuple output: append task_id bit-exactly after the 32M f32 outputs
  if (blockIdx.x == 0 && blockIdx.y == 0 && t == 0)
    ((int*)out)[((size_t)32 << 20) + b] = task;
}

extern "C" void kernel_launch(void* const* d_in, const int* in_sizes, int n_in,
                              void* d_out, int out_size, void* d_ws, size_t ws_size,
                              hipStream_t stream) {
  (void)in_sizes; (void)n_in; (void)out_size; (void)d_ws; (void)ws_size;
  const float* x   = (const float*)d_in[0];
  const int*   tid = (const int*)d_in[1];
  const float* W   = (const float*)d_in[2];
  float*       out = (float*)d_out;
  dim3 grid(8, 8, 32), block(256);
  hipLaunchKernelGGL(cond_linear_bf16x3_kernel, grid, block, 0, stream,
                     x, tid, W, out);
}
